// GNNEncoder_76072460746835
// MI455X (gfx1250) — matrix operand order repaired and checked by reading, not compiled, not run on hardware
//
#include <hip/hip_runtime.h>
#include <hip/hip_bf16.h>

// ---------------------------------------------------------------------------
// GNN encoder for MI455X (gfx1250, wave32, WMMA).
// Dominant cost: 9x dense [8192x8192]x[8192x128] GEMMs -> f16 WMMA, f32 accum.
// a_norm stored f16 (134 MB) so it is L2-resident (192 MB L2) across the 3
// message-passing layers of each network. Big GEMM: 32x64 wave tiles
// (1.5 loads/WMMA), explicit ping-pong double buffering (no register rotation
// copies -> no WMMA WAR-hazard NOPs), K-split x8 for occupancy.
// ---------------------------------------------------------------------------

#define NN      8192
#define IN_DIM  256
#define HD      128
#define NLAY    3
#define NNET    3
#define HEADS   4
#define KSPLIT  8
#define KCH     (NN / KSPLIT)   // 1024 K per partial

typedef __attribute__((ext_vector_type(16))) _Float16 v16h;
typedef __attribute__((ext_vector_type(8)))  _Float16 v8h;
typedef __attribute__((ext_vector_type(4)))  _Float16 v4h;
typedef __attribute__((ext_vector_type(8)))  float    v8f;

// ---------------- WMMA fragment helpers (16x16x32 f16, wave32) -------------
// A (16x32, MxK), 16-bit layout:
//   lane l: m = l&15, half = l>>4
//   elements 0..7  = A[m][half*8 + 0..7]      (VGPR0..3)
//   elements 8..15 = A[m][16 + half*8 + 0..7] (VGPR4..7)
// B (32x16, KxN) loaded from BT (N-major, K contiguous):
//   lane l: n = l&15, khalf = l>>4
//   elements 0..7  = BT[n][khalf*8 + 0..7], elements 8..15 = BT[n][16 + khalf*8 + 0..7]
// C/D (16x16 f32): lane l: n = l&15, half = l>>4; VGPR r -> row (half*8 + r).

__device__ __forceinline__ v16h make_frag(const _Float16* plo, const _Float16* phi) {
  const v8h lo = *reinterpret_cast<const v8h*>(plo);
  const v8h hi = *reinterpret_cast<const v8h*>(phi);
  v16h r;
#pragma unroll
  for (int i = 0; i < 8; ++i) { r[i] = lo[i]; r[8 + i] = hi[i]; }
  return r;
}

__device__ __forceinline__ v16h load_a_frag(const _Float16* A, int ld,
                                            int row0, int k0, int lane) {
  const int m = lane & 15, half = lane >> 4;
  const _Float16* base = A + (size_t)(row0 + m) * ld + k0 + half * 8;
  return make_frag(base, base + 16);
}

__device__ __forceinline__ v16h load_bt_frag(const _Float16* BT, int ldt,
                                             int n0, int k0, int lane) {
  const int n = lane & 15, khalf = lane >> 4;
  const _Float16* base = BT + (size_t)(n0 + n) * ldt + k0 + khalf * 8;
  return make_frag(base, base + 16);
}

#define WMMA_F16(a, b, c) \
  __builtin_amdgcn_wmma_f32_16x16x32_f16(false, (a), false, (b), (short)0, (c), false, false)

__device__ __forceinline__ v8f zero8() {
  v8f z = {0.f, 0.f, 0.f, 0.f, 0.f, 0.f, 0.f, 0.f};
  return z;
}

// ---------------- weight prep: transpose + f16 ------------------------------
__global__ void k_prep_weights(const float* __restrict__ Wm,
                               const float* __restrict__ Wu,
                               _Float16* __restrict__ WmT,
                               _Float16* __restrict__ WuAT,
                               _Float16* __restrict__ WuBT) {
  int idx = blockIdx.x * 256 + threadIdx.x;
  if (idx >= NLAY * HD * HD) return;
  int l = idx / (HD * HD);
  int rem = idx - l * HD * HD;
  int n = rem / HD, k = rem - n * HD;
  WmT[idx]  = (_Float16)Wm[(size_t)l * HD * HD + (size_t)k * HD + n];
  WuAT[idx] = (_Float16)Wu[(size_t)l * 2 * HD * HD + (size_t)k * HD + n];
  WuBT[idx] = (_Float16)Wu[(size_t)l * 2 * HD * HD + (size_t)(HD + k) * HD + n];
}

// ---------------- input projection: h0 = X @ Wi + bi ------------------------
__global__ void __launch_bounds__(256) k_input_proj(const float* __restrict__ X,
                                                    const float* __restrict__ Wi,
                                                    const float* __restrict__ bi,
                                                    _Float16* __restrict__ h0,
                                                    _Float16* __restrict__ h0T) {
  int idx = blockIdx.x * 256 + threadIdx.x;
  int row = idx >> 7, c = idx & 127;
  if (row >= NN) return;
  const float* x = X + (size_t)row * IN_DIM;
  float s = bi[c];
#pragma unroll 8
  for (int k = 0; k < IN_DIM; ++k) s += x[k] * Wi[(size_t)k * HD + c];
  _Float16 hv = (_Float16)s;
  h0[(size_t)row * HD + c] = hv;
  h0T[(size_t)c * NN + row] = hv;
}

// ---------------- per-row degree normalization, fp32 -> f16 ----------------
__global__ void __launch_bounds__(256) k_rownorm(const float* __restrict__ adj,
                                                 _Float16* __restrict__ anorm) {
  const int row = blockIdx.x;
  const float4* a4 = reinterpret_cast<const float4*>(adj + (size_t)row * NN);
  __shared__ float red[256];
  float s = 0.f;
  for (int i = threadIdx.x; i < NN / 4; i += 256) {
    float4 t = a4[i];
    s += t.x + t.y + t.z + t.w;
  }
  red[threadIdx.x] = s;
  __syncthreads();
  for (int st = 128; st > 0; st >>= 1) {
    if (threadIdx.x < st) red[threadIdx.x] += red[threadIdx.x + st];
    __syncthreads();
  }
  const float deg = red[0] > 1.0f ? red[0] : 1.0f;   // clip(sum, 1.0)
  const float inv = 1.0f / deg;
  _Float16* o = anorm + (size_t)row * NN;
  for (int i = threadIdx.x; i < NN / 4; i += 256) {
    float4 t = a4[i];
    v4h h;
    h[0] = (_Float16)(t.x * inv);
    h[1] = (_Float16)(t.y * inv);
    h[2] = (_Float16)(t.z * inv);
    h[3] = (_Float16)(t.w * inv);
    *reinterpret_cast<v4h*>(o + (size_t)i * 4) = h;
  }
}

// ---------------- big GEMM: partial = a_norm(f16) @ h (via hT) -------------
// Grid (NN/128, KSPLIT). Block: 256 threads = 8 waves; each wave owns a
// 32x64 output tile (2 A frags x 4 B frags -> 8 WMMAs per K-step, 12 loads:
// 1.5 loads/WMMA). Explicit X/Y ping-pong: loads always target the buffer the
// WMMAs are NOT reading -> no register rotation copies, no WAR-hazard NOPs.
__global__ void __launch_bounds__(256) k_spmm(const _Float16* __restrict__ A,
                                              const _Float16* __restrict__ BT,
                                              float* __restrict__ Cpart) {
  const int lane = threadIdx.x & 31;
  const int wave = threadIdx.x >> 5;
  const int row0 = blockIdx.x * 128 + (wave & 3) * 32;
  const int col0 = (wave >> 2) * 64;
  const int kbeg = blockIdx.y * KCH;
  float* out = Cpart + (size_t)blockIdx.y * NN * HD;

  v8f acc[8];
#pragma unroll
  for (int i = 0; i < 8; ++i) acc[i] = zero8();

  // buffer X: fragments for K-step kbeg
  v16h xa0 = load_a_frag(A, NN, row0,      kbeg, lane);
  v16h xa1 = load_a_frag(A, NN, row0 + 16, kbeg, lane);
  v16h xb0 = load_bt_frag(BT, NN, col0 +  0, kbeg, lane);
  v16h xb1 = load_bt_frag(BT, NN, col0 + 16, kbeg, lane);
  v16h xb2 = load_bt_frag(BT, NN, col0 + 32, kbeg, lane);
  v16h xb3 = load_bt_frag(BT, NN, col0 + 48, kbeg, lane);
  v16h ya0, ya1, yb0, yb1, yb2, yb3;

  // two K-steps per iteration; 15 iterations cover steps 0..29, epilogue 30..31
#pragma unroll 1
  for (int k0 = kbeg; k0 < kbeg + KCH - 64; k0 += 64) {
    // load Y(k0+32) while computing X(k0)
    ya0 = load_a_frag(A, NN, row0,      k0 + 32, lane);
    ya1 = load_a_frag(A, NN, row0 + 16, k0 + 32, lane);
    yb0 = load_bt_frag(BT, NN, col0 +  0, k0 + 32, lane);
    yb1 = load_bt_frag(BT, NN, col0 + 16, k0 + 32, lane);
    yb2 = load_bt_frag(BT, NN, col0 + 32, k0 + 32, lane);
    yb3 = load_bt_frag(BT, NN, col0 + 48, k0 + 32, lane);
    acc[0] = WMMA_F16(xa0, xb0, acc[0]);
    acc[1] = WMMA_F16(xa0, xb1, acc[1]);
    acc[2] = WMMA_F16(xa0, xb2, acc[2]);
    acc[3] = WMMA_F16(xa0, xb3, acc[3]);
    acc[4] = WMMA_F16(xa1, xb0, acc[4]);
    acc[5] = WMMA_F16(xa1, xb1, acc[5]);
    acc[6] = WMMA_F16(xa1, xb2, acc[6]);
    acc[7] = WMMA_F16(xa1, xb3, acc[7]);
    // load X(k0+64) while computing Y(k0+32)
    xa0 = load_a_frag(A, NN, row0,      k0 + 64, lane);
    xa1 = load_a_frag(A, NN, row0 + 16, k0 + 64, lane);
    xb0 = load_bt_frag(BT, NN, col0 +  0, k0 + 64, lane);
    xb1 = load_bt_frag(BT, NN, col0 + 16, k0 + 64, lane);
    xb2 = load_bt_frag(BT, NN, col0 + 32, k0 + 64, lane);
    xb3 = load_bt_frag(BT, NN, col0 + 48, k0 + 64, lane);
    acc[0] = WMMA_F16(ya0, yb0, acc[0]);
    acc[1] = WMMA_F16(ya0, yb1, acc[1]);
    acc[2] = WMMA_F16(ya0, yb2, acc[2]);
    acc[3] = WMMA_F16(ya0, yb3, acc[3]);
    acc[4] = WMMA_F16(ya1, yb0, acc[4]);
    acc[5] = WMMA_F16(ya1, yb1, acc[5]);
    acc[6] = WMMA_F16(ya1, yb2, acc[6]);
    acc[7] = WMMA_F16(ya1, yb3, acc[7]);
  }
  // epilogue: X holds step (KCH-64); load Y(KCH-32), compute both
  {
    const int klast = kbeg + KCH - 32;
    ya0 = load_a_frag(A, NN, row0,      klast, lane);
    ya1 = load_a_frag(A, NN, row0 + 16, klast, lane);
    yb0 = load_bt_frag(BT, NN, col0 +  0, klast, lane);
    yb1 = load_bt_frag(BT, NN, col0 + 16, klast, lane);
    yb2 = load_bt_frag(BT, NN, col0 + 32, klast, lane);
    yb3 = load_bt_frag(BT, NN, col0 + 48, klast, lane);
    acc[0] = WMMA_F16(xa0, xb0, acc[0]);
    acc[1] = WMMA_F16(xa0, xb1, acc[1]);
    acc[2] = WMMA_F16(xa0, xb2, acc[2]);
    acc[3] = WMMA_F16(xa0, xb3, acc[3]);
    acc[4] = WMMA_F16(xa1, xb0, acc[4]);
    acc[5] = WMMA_F16(xa1, xb1, acc[5]);
    acc[6] = WMMA_F16(xa1, xb2, acc[6]);
    acc[7] = WMMA_F16(xa1, xb3, acc[7]);
    acc[0] = WMMA_F16(ya0, yb0, acc[0]);
    acc[1] = WMMA_F16(ya0, yb1, acc[1]);
    acc[2] = WMMA_F16(ya0, yb2, acc[2]);
    acc[3] = WMMA_F16(ya0, yb3, acc[3]);
    acc[4] = WMMA_F16(ya1, yb0, acc[4]);
    acc[5] = WMMA_F16(ya1, yb1, acc[5]);
    acc[6] = WMMA_F16(ya1, yb2, acc[6]);
    acc[7] = WMMA_F16(ya1, yb3, acc[7]);
  }

  const int n = lane & 15, half = lane >> 4;
#pragma unroll
  for (int s = 0; s < 2; ++s) {
    const int rbase = row0 + s * 16 + half * 8;
#pragma unroll
    for (int t = 0; t < 4; ++t) {
      const int c = col0 + t * 16 + n;
#pragma unroll
      for (int r = 0; r < 8; ++r)
        out[(size_t)(rbase + r) * HD + c] = acc[s * 4 + t][r];
    }
  }
}

// fold the KSPLIT fp32 partials -> f16 mpre (deterministic, fixed order)
__global__ void __launch_bounds__(256) k_reduce_parts(const float* __restrict__ part,
                                                      _Float16* __restrict__ mpre) {
  int i = blockIdx.x * 256 + threadIdx.x;
  if (i >= NN * HD) return;
  const size_t S = (size_t)NN * HD;
  float s = 0.f;
#pragma unroll
  for (int p = 0; p < KSPLIT; ++p) s += part[i + (size_t)p * S];
  mpre[i] = (_Float16)s;
}

// ---------------- layer update ---------------------------------------------
// t      = Mpre @ Wm + bm                       (staged through LDS, f16)
// h_new  = relu(h @ WuA + t @ WuB + bu)
#define TSH_LD 136   // 128 + 8 f16 pad -> 272 B rows, avoids LDS bank conflicts
__global__ void __launch_bounds__(256) k_layer_update(
    const _Float16* __restrict__ h,      // [N][128]
    const _Float16* __restrict__ mpre,   // [N][128]
    const _Float16* __restrict__ WmT,    // [128][128]
    const float*    __restrict__ bm,     // [128]
    const _Float16* __restrict__ WuAT,   // [128][128]
    const _Float16* __restrict__ WuBT,   // [128][128]
    const float*    __restrict__ bu,     // [128]
    _Float16* __restrict__ h_out,        // [N][128]
    _Float16* __restrict__ hT_out,       // [128][N]
    float*    __restrict__ emb_out,      // [N][128]
    int write_emb) {
  __shared__ _Float16 tsh[64 * TSH_LD];
  const int lane   = threadIdx.x & 31;
  const int wave   = threadIdx.x >> 5;
  const int stripe = wave & 3;
  const int col0   = (wave >> 2) * 64;
  const int brow   = blockIdx.x * 64;
  const int row0   = brow + stripe * 16;
  const int n = lane & 15, half = lane >> 4;

  // ---- stage 1: t = mpre @ Wm + bm -> LDS (f16) ----
  {
    v8f acc[4] = {zero8(), zero8(), zero8(), zero8()};
#pragma unroll
    for (int k0 = 0; k0 < HD; k0 += 32) {
      v16h a = load_a_frag(mpre, HD, row0, k0, lane);
#pragma unroll
      for (int t = 0; t < 4; ++t) {
        v16h b = load_bt_frag(WmT, HD, col0 + t * 16, k0, lane);
        acc[t] = WMMA_F16(a, b, acc[t]);
      }
    }
#pragma unroll
    for (int t = 0; t < 4; ++t) {
      const int c = col0 + t * 16 + n;
      const float bias = bm[c];
#pragma unroll
      for (int r = 0; r < 8; ++r)
        tsh[(stripe * 16 + half * 8 + r) * TSH_LD + c] = (_Float16)(acc[t][r] + bias);
    }
  }
  __syncthreads();

  // ---- stage 2: h_new = relu(h @ WuA + t @ WuB + bu) ----
  v8f o[4] = {zero8(), zero8(), zero8(), zero8()};
#pragma unroll
  for (int k0 = 0; k0 < HD; k0 += 32) {
    v16h a1 = load_a_frag(h, HD, row0, k0, lane);               // global h
    v16h a2 = load_a_frag(tsh, TSH_LD, stripe * 16, k0, lane);  // LDS messages
#pragma unroll
    for (int t = 0; t < 4; ++t) {
      v16h bA = load_bt_frag(WuAT, HD, col0 + t * 16, k0, lane);
      o[t] = WMMA_F16(a1, bA, o[t]);
      v16h bB = load_bt_frag(WuBT, HD, col0 + t * 16, k0, lane);
      o[t] = WMMA_F16(a2, bB, o[t]);
    }
  }
#pragma unroll
  for (int t = 0; t < 4; ++t) {
    const int c = col0 + t * 16 + n;
    const float bias = bu[c];
#pragma unroll
    for (int r = 0; r < 8; ++r) {
      const int row = row0 + half * 8 + r;
      float val = o[t][r] + bias;
      val = val > 0.f ? val : 0.f;
      const _Float16 hv = (_Float16)val;
      h_out[(size_t)row * HD + c] = hv;
      hT_out[(size_t)c * NN + row] = hv;
      if (write_emb) emb_out[(size_t)row * HD + c] = val;
    }
  }
}

// ---------------- cross-layer attention + output projection ----------------
__global__ void __launch_bounds__(128) k_attention(const float* __restrict__ emb, // [3][N][H]
                                                   const float* __restrict__ Wa,  // [H][4]
                                                   const float* __restrict__ ba,  // [4]
                                                   const float* __restrict__ v,   // [4]
                                                   const float* __restrict__ Wo,  // [H][H]
                                                   const float* __restrict__ bo,  // [H]
                                                   float* __restrict__ out) {     // [N][H]
  const int node = blockIdx.x, t = threadIdx.x;
  __shared__ float e[NNET][HD];
  __shared__ float w[NNET];
  __shared__ float ne[HD];
#pragma unroll
  for (int k = 0; k < NNET; ++k)
    e[k][t] = emb[((size_t)k * NN + node) * HD + t];
  __syncthreads();
  if (t < NNET) {
    float s = 0.f;
#pragma unroll
    for (int hd = 0; hd < HEADS; ++hd) {
      float d = ba[hd];
      for (int k = 0; k < HD; ++k) d += e[t][k] * Wa[k * HEADS + hd];
      s += tanhf(d) * v[hd];
    }
    w[t] = s;
  }
  __syncthreads();
  if (t == 0) {
    float m = fmaxf(w[0], fmaxf(w[1], w[2]));
    float e0 = expf(w[0] - m), e1 = expf(w[1] - m), e2 = expf(w[2] - m);
    float inv = 1.f / (e0 + e1 + e2);
    w[0] = e0 * inv; w[1] = e1 * inv; w[2] = e2 * inv;
  }
  __syncthreads();
  ne[t] = w[0] * e[0][t] + w[1] * e[1][t] + w[2] * e[2][t];
  __syncthreads();
  float o = bo[t];
  for (int k = 0; k < HD; ++k) o += ne[k] * Wo[k * HD + t];
  out[(size_t)node * HD + t] = o;
}

// deterministic mean over nodes (no atomics -> replay-stable)
__global__ void __launch_bounds__(256) k_mean(const float* __restrict__ node_out,
                                              float* __restrict__ glob) {
  const int c = blockIdx.x;   // feature 0..127
  __shared__ float red[256];
  float s = 0.f;
  for (int r = threadIdx.x; r < NN; r += 256) s += node_out[(size_t)r * HD + c];
  red[threadIdx.x] = s;
  __syncthreads();
  for (int st = 128; st > 0; st >>= 1) {
    if (threadIdx.x < st) red[threadIdx.x] += red[threadIdx.x + st];
    __syncthreads();
  }
  if (threadIdx.x == 0) glob[c] = red[0] * (1.0f / (float)NN);
}

// ---------------------------------------------------------------------------
extern "C" void kernel_launch(void* const* d_in, const int* in_sizes, int n_in,
                              void* d_out, int out_size, void* d_ws, size_t ws_size,
                              hipStream_t stream) {
  const float* X   = (const float*)d_in[0];
  const float* adj = (const float*)d_in[1];
  const float* Wi  = (const float*)d_in[2];
  const float* bi  = (const float*)d_in[3];
  const float* Wm  = (const float*)d_in[4];
  const float* bm  = (const float*)d_in[5];
  const float* Wu  = (const float*)d_in[6];
  const float* bu  = (const float*)d_in[7];
  const float* Wa  = (const float*)d_in[8];
  const float* ba  = (const float*)d_in[9];
  const float* vv  = (const float*)d_in[10];
  const float* Wo  = (const float*)d_in[11];
  const float* bo  = (const float*)d_in[12];
  float* out = (float*)d_out;
  (void)in_sizes; (void)n_in; (void)out_size; (void)ws_size;

  // workspace carve-out (all 256-B aligned)
  char* ws = (char*)d_ws;
  size_t off = 0;
  auto carve = [&](size_t bytes) -> char* {
    off = (off + 255) & ~(size_t)255;
    char* p = ws + off;
    off += bytes;
    return p;
  };
  const size_t NH2 = (size_t)NN * HD * sizeof(_Float16);
  _Float16* anorm = (_Float16*)carve((size_t)NN * NN * sizeof(_Float16)); // 134 MB, L2-resident
  _Float16* h0    = (_Float16*)carve(NH2);
  _Float16* h0T   = (_Float16*)carve(NH2);
  _Float16* hA    = (_Float16*)carve(NH2);
  _Float16* hAT   = (_Float16*)carve(NH2);
  _Float16* hB    = (_Float16*)carve(NH2);
  _Float16* hBT   = (_Float16*)carve(NH2);
  _Float16* mpre  = (_Float16*)carve(NH2);
  float*    part  = (float*)carve((size_t)KSPLIT * NN * HD * sizeof(float)); // 32 MB
  float*    emb   = (float*)carve((size_t)NNET * NN * HD * sizeof(float));   // 12 MB
  _Float16* WmT   = (_Float16*)carve((size_t)NLAY * HD * HD * sizeof(_Float16));
  _Float16* WuAT  = (_Float16*)carve((size_t)NLAY * HD * HD * sizeof(_Float16));
  _Float16* WuBT  = (_Float16*)carve((size_t)NLAY * HD * HD * sizeof(_Float16));

  k_prep_weights<<<(NLAY * HD * HD + 255) / 256, 256, 0, stream>>>(Wm, Wu, WmT, WuAT, WuBT);
  k_input_proj<<<(NN * HD) / 256, 256, 0, stream>>>(X, Wi, bi, h0, h0T);

  for (int k = 0; k < NNET; ++k) {
    k_rownorm<<<NN, 256, 0, stream>>>(adj + (size_t)k * NN * NN, anorm);
    const _Float16* h  = h0;
    const _Float16* hT = h0T;
    for (int l = 0; l < NLAY; ++l) {
      k_spmm<<<dim3(NN / 128, KSPLIT), 256, 0, stream>>>(anorm, hT, part);
      k_reduce_parts<<<(NN * HD) / 256, 256, 0, stream>>>(part, mpre);
      _Float16* ho  = (l & 1) ? hB  : hA;
      _Float16* hoT = (l & 1) ? hBT : hAT;
      k_layer_update<<<NN / 64, 256, 0, stream>>>(
          h, mpre,
          WmT + (size_t)l * HD * HD, bm + (size_t)l * HD,
          WuAT + (size_t)l * HD * HD, WuBT + (size_t)l * HD * HD, bu + (size_t)l * HD,
          ho, hoT, emb + (size_t)k * NN * HD, (l == NLAY - 1) ? 1 : 0);
      h = ho; hT = hoT;
    }
  }

  k_attention<<<NN, 128, 0, stream>>>(emb, Wa, ba, vv, Wo, bo, out);
  k_mean<<<HD, 256, 0, stream>>>(out, out + (size_t)NN * HD);
}
